// FP8Linear_90872918048786
// MI455X (gfx1250) — 8-bit matrix operand order repaired and checked by reading, compiled, not run on hardware
//
#include <hip/hip_runtime.h>
#include <hip/hip_bf16.h>

typedef __attribute__((ext_vector_type(16))) int    v16i;
typedef __attribute__((ext_vector_type(8)))  float  v8f;
typedef __attribute__((ext_vector_type(4)))  float  v4f;

#define KDIM 256
#define NDIM 256
#define TM   128
#define QBLK 128

#if defined(__has_builtin)
#if __has_builtin(__builtin_amdgcn_cvt_pk_fp8_f32)
#define HAVE_HW_FP8 1
#endif
#endif

__device__ __forceinline__ unsigned int e4m3_byte(float q) {
    unsigned int sign = (__float_as_uint(q) >> 24) & 0x80u;
    int ia = (int)fabsf(q);
    if (ia == 0) return sign;
    int e = 31 - __clz(ia);
    int m4;
    if (e <= 3) {
        m4 = ia << (3 - e);
    } else {
        int sh   = e - 3;
        int base = ia >> sh;
        int rem  = ia & ((1 << sh) - 1);
        int half = 1 << (sh - 1);
        if (rem > half || (rem == half && (base & 1))) base++;
        if (base == 16) { base >>= 1; e++; }
        m4 = base;
    }
    return sign | ((unsigned)(e + 7) << 3) | (unsigned)(m4 & 7);
}

__device__ __forceinline__ unsigned int pack4_e4m3(float q0, float q1, float q2, float q3) {
#ifdef HAVE_HW_FP8
    int p = __builtin_amdgcn_cvt_pk_fp8_f32(q0, q1, 0, false);
    p     = __builtin_amdgcn_cvt_pk_fp8_f32(q2, q3, p, true);
    return (unsigned int)p;
#else
    return e4m3_byte(q0) | (e4m3_byte(q1) << 8) | (e4m3_byte(q2) << 16) | (e4m3_byte(q3) << 24);
#endif
}

__device__ __forceinline__ unsigned int one_e4m3(float q) {
#ifdef HAVE_HW_FP8
    return (unsigned int)__builtin_amdgcn_cvt_pk_fp8_f32(q, 0.0f, 0, false) & 0xffu;
#else
    return e4m3_byte(q);
#endif
}

__global__ __launch_bounds__(256) void qweight_kernel(const float* __restrict__ w,
                                                      unsigned char* __restrict__ wq,
                                                      float* __restrict__ rs) {
    __shared__ unsigned int amax[4];
    const int t = threadIdx.x;
    if (t < 4) amax[t] = 0u;
    __syncthreads();

    float m0 = 0.f, m1 = 0.f;
    for (int k = 0; k < QBLK; ++k) {
        m0 = fmaxf(m0, fabsf(w[t * KDIM + k]));
        m1 = fmaxf(m1, fabsf(w[t * KDIM + QBLK + k]));
    }
    const int ob = t >> 7;
    atomicMax(&amax[ob * 2 + 0], __float_as_uint(m0));
    atomicMax(&amax[ob * 2 + 1], __float_as_uint(m1));
    __syncthreads();

    if (t < 4) {
        unsigned int u = amax[t];
        rs[t] = (u == 0u) ? 1.0f : __uint_as_float(u);
    }

    float s[2];
    s[0] = (amax[ob * 2 + 0] == 0u) ? 1.0f : __uint_as_float(amax[ob * 2 + 0]);
    s[1] = (amax[ob * 2 + 1] == 0u) ? 1.0f : __uint_as_float(amax[ob * 2 + 1]);

    const int ntile = t >> 4;
    const int n     = t & 15;
    for (int k = 0; k < KDIM; ++k) {
        const int kb = k >> 7;
        const int kl = k & 127;
        float q = rintf(w[t * KDIM + k] * (127.0f / s[kb]));
        q = fminf(fmaxf(q, -127.0f), 127.0f);
        const int g    = kl >> 5;
        const int kidx = kl & 31;
        const int lane = (kidx < 16) ? n : (16 + n);
        const int dw   = g * 4 + ((kidx & 15) >> 2);
        wq[(((ntile * 2 + kb) * 32 + lane) << 6) + dw * 4 + (kidx & 3)] = (unsigned char)one_e4m3(q);
    }
}

__global__ __launch_bounds__(256) void fp8gemm_kernel(const float* __restrict__ x,
                                                      const unsigned char* __restrict__ wq,
                                                      const float* __restrict__ rs,
                                                      const float* __restrict__ bias,
                                                      float* __restrict__ out) {
    __shared__ unsigned char qlds[TM * KDIM];
    __shared__ float         slds[TM * 2];

    const int t   = threadIdx.x;
    const size_t row0 = (size_t)blockIdx.x * TM;

    {
        const int r  = t >> 1;
        const int kb = t & 1;
        const v4f* xp = (const v4f*)(x + (row0 + r) * KDIM + kb * QBLK);

        float mx = 0.f;
        #pragma unroll
        for (int j = 0; j < QBLK / 4; ++j) {
            const v4f v = xp[j];
            mx = fmaxf(mx, fmaxf(fmaxf(fabsf(v.x), fabsf(v.y)), fmaxf(fabsf(v.z), fabsf(v.w))));
        }
        const float sc = (mx == 0.f) ? 1.0f : mx;
        slds[r * 2 + kb] = sc;
        const float inv = 127.0f / sc;

        #pragma unroll
        for (int j = 0; j < QBLK / 4; ++j) {
            const v4f v = __builtin_nontemporal_load(&xp[j]);
            float q0 = fminf(fmaxf(rintf(v.x * inv), -127.f), 127.f);
            float q1 = fminf(fmaxf(rintf(v.y * inv), -127.f), 127.f);
            float q2 = fminf(fmaxf(rintf(v.z * inv), -127.f), 127.f);
            float q3 = fminf(fmaxf(rintf(v.w * inv), -127.f), 127.f);
            *(unsigned int*)&qlds[r * KDIM + kb * QBLK + j * 4] = pack4_e4m3(q0, q1, q2, q3);
        }
    }
    __syncthreads();

    const int lane = t & 31;
    const int wv   = t >> 5;
    const int m    = lane & 15;
    const int hl   = lane >> 4;

    v16i a0, a1;
    {
        const int rowl = wv * 16 + m;
        #pragma unroll
        for (int pr = 0; pr < 8; ++pr) {
            const int sI = pr >> 1;
            const int cI = pr & 1;
            const int k0 = sI * 32 + hl * 16 + cI * 8;
            const uint2 u0 = *(const uint2*)&qlds[rowl * KDIM + k0];
            const uint2 u1 = *(const uint2*)&qlds[rowl * KDIM + QBLK + k0];
            a0[pr * 2 + 0] = (int)u0.x; a0[pr * 2 + 1] = (int)u0.y;
            a1[pr * 2 + 0] = (int)u1.x; a1[pr * 2 + 1] = (int)u1.y;
        }
    }

    float ls0[8], ls1[8];
    #pragma unroll
    for (int i = 0; i < 8; ++i) {
        const int ml = wv * 16 + hl * 8 + i;
        ls0[i] = slds[ml * 2 + 0];
        ls1[i] = slds[ml * 2 + 1];
    }

    const float INV = 1.0f / (127.0f * 127.0f);

    auto loadB = [&](int nt, v16i& b0, v16i& b1) {
        const int4* bp0 = (const int4*)(wq + (((nt * 2 + 0) * 32 + lane) << 6));
        const int4* bp1 = (const int4*)(wq + (((nt * 2 + 1) * 32 + lane) << 6));
        #pragma unroll
        for (int j = 0; j < 4; ++j) {
            const int4 u0 = bp0[j], u1 = bp1[j];
            b0[j*4+0]=u0.x; b0[j*4+1]=u0.y; b0[j*4+2]=u0.z; b0[j*4+3]=u0.w;
            b1[j*4+0]=u1.x; b1[j*4+1]=u1.y; b1[j*4+2]=u1.z; b1[j*4+3]=u1.w;
        }
    };

    auto tileOut = [&](int nt, const v16i& b0, const v16i& b1, float r0, float r1) {
        v8f acc0 = {}, acc1 = {};
        acc0 = __builtin_amdgcn_wmma_f32_16x16x128_fp8_fp8(a0, b0, (short)0, acc0, false, false);
        acc1 = __builtin_amdgcn_wmma_f32_16x16x128_fp8_fp8(a1, b1, (short)0, acc1, false, false);

        const int col  = nt * 16 + m;
        const float bv = bias[col];

        #pragma unroll
        for (int i = 0; i < 8; ++i) {
            const int mlocal = wv * 16 + hl * 8 + i;
            float v = acc0[i] * (ls0[i] * r0) + acc1[i] * (ls1[i] * r1);
            v = __bfloat162float(__float2bfloat16(v));
            __builtin_nontemporal_store(v + bv, &out[(row0 + mlocal) * NDIM + col]);
        }
    };

    v16i bc0, bc1, bn0, bn1;
    loadB(0, bc0, bc1);

    #pragma unroll 1
    for (int ob = 0; ob < 2; ++ob) {
        const float r0 = rs[ob * 2 + 0] * INV;
        const float r1 = rs[ob * 2 + 1] * INV;
        #pragma unroll 1
        for (int it = 0; it < 4; ++it) {
            const int nt = ob * 8 + it * 2;
            loadB(nt + 1, bn0, bn1);
            tileOut(nt, bc0, bc1, r0, r1);
            loadB((nt + 2) & 15, bc0, bc1);
            tileOut(nt + 1, bn0, bn1, r0, r1);
        }
    }
}

extern "C" void kernel_launch(void* const* d_in, const int* in_sizes, int n_in,
                              void* d_out, int out_size, void* d_ws, size_t ws_size,
                              hipStream_t stream) {
    const float* x      = (const float*)d_in[0];
    const float* weight = (const float*)d_in[1];
    const float* bias   = (const float*)d_in[2];
    float* out          = (float*)d_out;

    unsigned char* wq = (unsigned char*)d_ws;
    float* rsp        = (float*)((char*)d_ws + 65536);

    const int M = in_sizes[0] / KDIM;

    qweight_kernel<<<1, 256, 0, stream>>>(weight, wq, rsp);
    fp8gemm_kernel<<<M / TM, 256, 0, stream>>>(x, wq, rsp, bias, out);
}
